// OrthoLinear_86251533238842
// MI455X (gfx1250) — matrix operand order repaired and checked by reading, not compiled, not run on hardware
//
#include <hip/hip_runtime.h>
#include <hip/hip_fp16.h>

#define IN_F   4096
#define OUT_F  4096
#define TOKENS 2048
#define ALPHA_F 1.0f

#define BM 128
#define BN 64

// fast path (pre-converted bf16 operands)
#define BK  64
#define BKP (BK + 8)        // 144B rows: 16B aligned, spreads banks
#define NIT (IN_F / BK)     // 64 K-steps

// fused fallback path
#define FBK  32
#define FBKP (FBK + 8)
#define FNIT (IN_F / FBK)

typedef __attribute__((ext_vector_type(16))) __bf16 v16bf;
typedef __attribute__((ext_vector_type(8)))  __bf16 v8bf;
typedef __attribute__((ext_vector_type(8)))  float  v8f;
typedef __attribute__((ext_vector_type(8)))  unsigned short u16x8;
typedef __attribute__((ext_vector_type(4)))  int    v4i;

__device__ __forceinline__ unsigned short f32_to_bf16_rne(float f) {
    union { float f; unsigned u; } v; v.f = f;
    unsigned b = v.u + 0x7FFFu + ((v.u >> 16) & 1u);
    return (unsigned short)(b >> 16);
}

__device__ __forceinline__ u16x8 pack_bf16x8(float4 a, float4 b) {
    u16x8 r;
    r[0] = f32_to_bf16_rne(a.x); r[1] = f32_to_bf16_rne(a.y);
    r[2] = f32_to_bf16_rne(a.z); r[3] = f32_to_bf16_rne(a.w);
    r[4] = f32_to_bf16_rne(b.x); r[5] = f32_to_bf16_rne(b.y);
    r[6] = f32_to_bf16_rne(b.z); r[7] = f32_to_bf16_rne(b.w);
    return r;
}

// 4 packed bytes (8 int4 codes, offset-by-8) -> 8 dequantized bf16
__device__ __forceinline__ u16x8 dequant8(unsigned pb, float s) {
    u16x8 r;
    #pragma unroll
    for (int i = 0; i < 4; ++i) {
        const int lo = (int)((pb >> (8 * i))     & 0xF) - 8;
        const int hi = (int)((pb >> (8 * i + 4)) & 0xF) - 8;
        r[2 * i]     = f32_to_bf16_rne((float)lo * s);
        r[2 * i + 1] = f32_to_bf16_rne((float)hi * s);
    }
    return r;
}

union Frag { v16bf v; v8bf h[2]; };

// ---- async global -> LDS staging (CDNA5 ASYNCcnt path), with fallback ----
#if __has_builtin(__builtin_amdgcn_global_load_async_to_lds_b128)
typedef __attribute__((address_space(1))) v4i GV4;   // int4 in global AS
typedef __attribute__((address_space(3))) v4i LV4;   // int4 in LDS AS
__device__ __forceinline__ void async_cp16(void* lds, const void* g) {
    __builtin_amdgcn_global_load_async_to_lds_b128((GV4*)g, (LV4*)lds, 0, 0);
}
__device__ __forceinline__ void async_wait0() {
#if __has_builtin(__builtin_amdgcn_s_wait_asynccnt)
    __builtin_amdgcn_s_wait_asynccnt(0);
#else
    asm volatile("s_wait_asynccnt 0x0" ::: "memory");
#endif
}
#else
__device__ __forceinline__ void async_cp16(void* lds, const void* g) {
    *(u16x8*)lds = *(const u16x8*)g;    // global_load_b128 + ds_store_b128
}
__device__ __forceinline__ void async_wait0() {}
#endif

__device__ __forceinline__ v8f wmma_bf16(const Frag& a, const Frag& b, v8f c) {
    return __builtin_amdgcn_wmma_f32_16x16x32_bf16(
               false, a.v, false, b.v, (short)0, c, false, false);
}

// -------------------------------------------------------------------------
// Pre-pass 1: x (f32) -> bf16, done ONCE (fused GEMM would redo it 64x).
// -------------------------------------------------------------------------
__global__ __launch_bounds__(256)
void cvt_x_bf16(const float* __restrict__ x, unsigned short* __restrict__ xbf) {
    const int i = (blockIdx.x * 256 + threadIdx.x) * 8;
    const float4 a = *(const float4*)(x + i);
    const float4 b = *(const float4*)(x + i + 4);
    *(u16x8*)(xbf + i) = pack_bf16x8(a, b);
}

// -------------------------------------------------------------------------
// Pre-pass 2: packed int4 W -> dequantized bf16, done ONCE (vs 16x fused).
// -------------------------------------------------------------------------
__global__ __launch_bounds__(256)
void dequant_w_bf16(const unsigned char* __restrict__ wp,
                    const float* __restrict__ scales,
                    unsigned short* __restrict__ wbf) {
    const int t = blockIdx.x * 256 + threadIdx.x;   // one 4-byte packed unit
    const int byteoff = t * 4;
    const int row = byteoff >> 11;                  // / (IN_F/2)
    const unsigned pb = *(const unsigned*)(wp + byteoff);
    *(u16x8*)(wbf + byteoff * 2) = dequant8(pb, scales[row]);
}

// -------------------------------------------------------------------------
// Fast GEMM: pure bf16 WMMA. 128x64 block tile, 8 wave32, each wave a
// 32x32 register tile -> 8x V_WMMA_F32_16X16X32_BF16 per 64-wide K-step.
// Tiles staged via async global->LDS copies (no VGPR round-trip, no
// conversion VALU); double-buffered, one s_wait_asynccnt + barrier/step.
// -------------------------------------------------------------------------
__global__ __launch_bounds__(256)
void ortho_gemm_bf16(const unsigned short* __restrict__ xbf,  // [TOKENS, IN_F]
                     const unsigned short* __restrict__ wbf,  // [OUT_F, IN_F]
                     float* __restrict__ y) {                 // [TOKENS, OUT_F]
    __shared__ unsigned short lds_x[2][BM][BKP];  // 2 x 18KB
    __shared__ unsigned short lds_w[2][BN][BKP];  // 2 x  9KB

    const int m0 = blockIdx.x * BM, n0 = blockIdx.y * BN;
    const int tid = threadIdx.x, lane = tid & 31, wave = tid >> 5;
    const int wm = wave & 3, wn = wave >> 2;
    const int half = lane >> 4, l15 = lane & 15;

    const int xr = tid >> 1, xc = (tid & 1) * 32;   // 32 bf16 (64B)/thread
    const int wr = tid >> 2, wc = (tid & 3) * 16;   // 16 bf16 (32B)/thread

    const unsigned short* gx = xbf + (m0 + xr) * IN_F + xc;
    const unsigned short* gw = wbf + (n0 + wr) * IN_F + wc;

    v8f acc00 = {}, acc01 = {}, acc10 = {}, acc11 = {};

    // ---- prologue: stage slab 0 into buffer 0 ----
    #pragma unroll
    for (int j = 0; j < 4; ++j) async_cp16(&lds_x[0][xr][xc + 8 * j], gx + 8 * j);
    #pragma unroll
    for (int j = 0; j < 2; ++j) async_cp16(&lds_w[0][wr][wc + 8 * j], gw + 8 * j);
    async_wait0();
    __syncthreads();

    for (int it = 0; it < NIT; ++it) {
        const int cur = it & 1, nxt = cur ^ 1;

        // async-stage slab it+1 into `nxt`; copies run behind the WMMAs
        if (it + 1 < NIT) {
            const int k = (it + 1) * BK;
            #pragma unroll
            for (int j = 0; j < 4; ++j)
                async_cp16(&lds_x[nxt][xr][xc + 8 * j], gx + k + 8 * j);
            #pragma unroll
            for (int j = 0; j < 2; ++j)
                async_cp16(&lds_w[nxt][wr][wc + 8 * j], gw + k + 8 * j);
            if (it + 2 < NIT) {
                __builtin_prefetch(gx + (it + 2) * BK, 0, 1);
                __builtin_prefetch(gw + (it + 2) * BK, 0, 1);
            }
        }

        // two K=32 chunks from buffer `cur`
        #pragma unroll
        for (int c = 0; c < 2; ++c) {
            Frag a0, a1, b0, b1;
            // A (16x32): lanes 0-15 M=lane K=0..7,16..23; 16-31 M=l-16 K=8..15,24..31
            const unsigned short* ar0 = &lds_x[cur][wm * 32 + l15][c * 32];
            const unsigned short* ar1 = &lds_x[cur][wm * 32 + 16 + l15][c * 32];
            a0.h[0] = *(const v8bf*)(ar0 + half * 8);
            a0.h[1] = *(const v8bf*)(ar0 + half * 8 + 16);
            a1.h[0] = *(const v8bf*)(ar1 + half * 8);
            a1.h[1] = *(const v8bf*)(ar1 + half * 8 + 16);
            // B (32x16): lane = N col; lanes 0-15 K=0..7,16..23; 16-31 K=8..15,24..31
            const unsigned short* br0 = &lds_w[cur][wn * 32 + l15][c * 32 + half * 8];
            const unsigned short* br1 = &lds_w[cur][wn * 32 + 16 + l15][c * 32 + half * 8];
            b0.h[0] = *(const v8bf*)(br0);
            b0.h[1] = *(const v8bf*)(br0 + 16);
            b1.h[0] = *(const v8bf*)(br1);
            b1.h[1] = *(const v8bf*)(br1 + 16);

            acc00 = wmma_bf16(a0, b0, acc00);
            acc01 = wmma_bf16(a0, b1, acc01);
            acc10 = wmma_bf16(a1, b0, acc10);
            acc11 = wmma_bf16(a1, b1, acc11);
        }

        async_wait0();
        __syncthreads();
    }

    // C/D layout: VGPR j -> M = j + 8*half, N = l15
    const int row0 = m0 + wm * 32 + half * 8;
    const int col  = n0 + wn * 32 + l15;
    #pragma unroll
    for (int j = 0; j < 8; ++j) {
        y[(row0 + j) * OUT_F + col]           = acc00[j];
        y[(row0 + j) * OUT_F + col + 16]      = acc01[j];
        y[(row0 + j + 16) * OUT_F + col]      = acc10[j];
        y[(row0 + j + 16) * OUT_F + col + 16] = acc11[j];
    }
}

// -------------------------------------------------------------------------
// Fallback fused GEMM (used only if d_ws can't hold the bf16 operands).
// -------------------------------------------------------------------------
__global__ __launch_bounds__(256)
void ortho_gemm_fused(const float* __restrict__ x,
                      const unsigned char* __restrict__ wp,
                      const float* __restrict__ scales,
                      float* __restrict__ y) {
    __shared__ unsigned short lds_x[2][BM][FBKP];
    __shared__ unsigned short lds_w[2][BN][FBKP];

    const int m0 = blockIdx.x * BM, n0 = blockIdx.y * BN;
    const int tid = threadIdx.x, lane = tid & 31, wave = tid >> 5;
    const int wm = wave & 3, wn = wave >> 2;
    const int half = lane >> 4, l15 = lane & 15;

    const int xr = tid >> 1, xc = (tid & 1) * 16;
    const int wr = tid >> 2, wc = (tid & 3) * 4;
    const float wscale = scales[n0 + wr];

    const float*         xg = x  + (m0 + xr) * IN_F + xc;
    const unsigned char* wg = wp + (n0 + wr) * (IN_F / 2) + wc;

    v8f acc00 = {}, acc01 = {}, acc10 = {}, acc11 = {};

    float4 xa0 = *(const float4*)(xg + 0), xa1 = *(const float4*)(xg + 4);
    float4 xa2 = *(const float4*)(xg + 8), xa3 = *(const float4*)(xg + 12);
    unsigned pb = *(const unsigned*)wg;
    *(u16x8*)&lds_x[0][xr][xc]     = pack_bf16x8(xa0, xa1);
    *(u16x8*)&lds_x[0][xr][xc + 8] = pack_bf16x8(xa2, xa3);
    *(u16x8*)&lds_w[0][wr][wc * 2] = dequant8(pb, wscale);
    __syncthreads();

    for (int it = 0; it < FNIT; ++it) {
        const int cur = it & 1, nxt = cur ^ 1;
        const bool more = (it + 1) < FNIT;
        if (more) {
            const int k = (it + 1) * FBK;
            xa0 = *(const float4*)(xg + k + 0);  xa1 = *(const float4*)(xg + k + 4);
            xa2 = *(const float4*)(xg + k + 8);  xa3 = *(const float4*)(xg + k + 12);
            pb  = *(const unsigned*)(wg + k / 2);
        }
        Frag a0, a1, b0, b1;
        const unsigned short* ar0 = &lds_x[cur][wm * 32 + l15][0];
        const unsigned short* ar1 = &lds_x[cur][wm * 32 + 16 + l15][0];
        a0.h[0] = *(const v8bf*)(ar0 + half * 8);
        a0.h[1] = *(const v8bf*)(ar0 + half * 8 + 16);
        a1.h[0] = *(const v8bf*)(ar1 + half * 8);
        a1.h[1] = *(const v8bf*)(ar1 + half * 8 + 16);
        const unsigned short* br0 = &lds_w[cur][wn * 32 + l15][half * 8];
        const unsigned short* br1 = &lds_w[cur][wn * 32 + 16 + l15][half * 8];
        b0.h[0] = *(const v8bf*)(br0); b0.h[1] = *(const v8bf*)(br0 + 16);
        b1.h[0] = *(const v8bf*)(br1); b1.h[1] = *(const v8bf*)(br1 + 16);

        acc00 = wmma_bf16(a0, b0, acc00);
        acc01 = wmma_bf16(a0, b1, acc01);
        acc10 = wmma_bf16(a1, b0, acc10);
        acc11 = wmma_bf16(a1, b1, acc11);

        if (more) {
            *(u16x8*)&lds_x[nxt][xr][xc]     = pack_bf16x8(xa0, xa1);
            *(u16x8*)&lds_x[nxt][xr][xc + 8] = pack_bf16x8(xa2, xa3);
            *(u16x8*)&lds_w[nxt][wr][wc * 2] = dequant8(pb, wscale);
        }
        __syncthreads();
    }

    const int row0 = m0 + wm * 32 + half * 8;
    const int col  = n0 + wn * 32 + l15;
    #pragma unroll
    for (int j = 0; j < 8; ++j) {
        y[(row0 + j) * OUT_F + col]           = acc00[j];
        y[(row0 + j) * OUT_F + col + 16]      = acc01[j];
        y[(row0 + j + 16) * OUT_F + col]      = acc10[j];
        y[(row0 + j + 16) * OUT_F + col + 16] = acc11[j];
    }
}

// -------------------------------------------------------------------------
// CSR outlier correction: y[t, r] += alpha * sum_i vals[i] * x[t, col[i]]
// val/col wave-uniform (scalar loads); x gathers L2-resident (32MB << 192MB).
// -------------------------------------------------------------------------
__global__ __launch_bounds__(256)
void ortho_sparse_csr(const float* __restrict__ x,
                      const _Float16* __restrict__ vals,
                      const int* __restrict__ cols,
                      const int* __restrict__ ptr,
                      float* __restrict__ y) {
    const int r = blockIdx.x;
    const int t = blockIdx.y * 256 + threadIdx.x;
    const int beg = ptr[r], end = ptr[r + 1];
    const float* xrow = x + t * IN_F;
    float acc = 0.0f;
    for (int i = beg; i < end; ++i) {
        acc += (float)vals[i] * xrow[cols[i]];
    }
    y[t * OUT_F + r] += ALPHA_F * acc;
}

extern "C" void kernel_launch(void* const* d_in, const int* in_sizes, int n_in,
                              void* d_out, int out_size, void* d_ws, size_t ws_size,
                              hipStream_t stream) {
    (void)in_sizes; (void)n_in; (void)out_size;
    const float*         x      = (const float*)d_in[0];
    const unsigned char* wp     = (const unsigned char*)d_in[1];
    const float*         scales = (const float*)d_in[2];
    const _Float16*      vals   = (const _Float16*)d_in[3];
    const int*           cols   = (const int*)d_in[4];
    const int*           ptr    = (const int*)d_in[5];
    float*               y      = (float*)d_out;

    const size_t xbf_elems = (size_t)TOKENS * IN_F;
    const size_t wbf_elems = (size_t)OUT_F * IN_F;
    const size_t need      = (xbf_elems + wbf_elems) * sizeof(unsigned short);

    dim3 gg(TOKENS / BM, OUT_F / BN);   // (16, 64)

    if (ws_size >= need) {
        unsigned short* xbf = (unsigned short*)d_ws;
        unsigned short* wbf = xbf + xbf_elems;
        cvt_x_bf16<<<(int)(xbf_elems / 8 / 256), 256, 0, stream>>>(x, xbf);
        dequant_w_bf16<<<(int)(wbf_elems / 8 / 256), 256, 0, stream>>>(wp, scales, wbf);
        ortho_gemm_bf16<<<gg, 256, 0, stream>>>(xbf, wbf, y);
    } else {
        ortho_gemm_fused<<<gg, 256, 0, stream>>>(x, wp, scales, y);
    }

    dim3 gs(OUT_F, TOKENS / 256);       // (4096, 8)
    ortho_sparse_csr<<<gs, 256, 0, stream>>>(x, vals, cols, ptr, y);
}
